// ROIPooler_35175782154298
// MI455X (gfx1250) — matrix operand order repaired and checked by reading, not compiled, not run on hardware
//
#include <hip/hip_runtime.h>
#include <math.h>

// ---------------------------------------------------------------------------
// FPN ROIAlign for MI455X (gfx1250).
// Separable formulation: out[c][by][bx] = sum_y sum_x F[c][y][x]*Ay[y][by]*Ax[x][bx]
//   Stage1 (VALU): y-interp+pool into LDS tile G[7][16][64] (coalesced row reads;
//                  features fit in 192MB L2 so these are L2 hits).
//   Stage2 (WMMA): x-interp as 16x16x32 f16 matmul with hi/lo error-compensated
//                  split (AhBh + AhBl + AlBh) for ~f32 accuracy.
//   Wave 7 prefetches the next channel tile (global_prefetch_b8) during stage2.
// ---------------------------------------------------------------------------

typedef __attribute__((ext_vector_type(16))) _Float16 v16h;
typedef __attribute__((ext_vector_type(8)))  _Float16 v8h;
typedef __attribute__((ext_vector_type(8)))  float    v8f;

#define OUTS  7
#define BMAX  128   // max x-band (level assignment bounds feature-space extent <= ~104)
#define SLAB  64    // x-band slab streamed through LDS per step (2 K-chunks of 32)

__global__ __launch_bounds__(256) void roi_align_fpn_kernel(
    const float* __restrict__ f0, const float* __restrict__ f1,
    const float* __restrict__ f2, const float* __restrict__ f3,
    const float* __restrict__ boxes, float* __restrict__ out)
{
    __shared__ __attribute__((aligned(16))) _Float16 sGh[7 * 16 * SLAB];
    __shared__ __attribute__((aligned(16))) _Float16 sGl[7 * 16 * SLAB];
    __shared__ __attribute__((aligned(16))) _Float16 sAxh[16 * BMAX];
    __shared__ __attribute__((aligned(16))) _Float16 sAxl[16 * BMAX];
    __shared__ int   sRy[7][4];
    __shared__ float sWy[7][4];
    __shared__ int   sXi[14][2];
    __shared__ float sXw[14][2];
    __shared__ int   sI[8];  // l, H, W, HW, xmin, Bclamped, nchunk, nslab

    const int tid = threadIdx.x;
    const int roi = blockIdx.x;
    const int nb  = roi >> 9;  // 512 ROIs per batch image

    // ---------------- per-ROI setup (single thread; tiny) ----------------
    if (tid == 0) {
        const float* bp = boxes + (size_t)roi * 4;
        float x1b = bp[0], y1b = bp[1], x2b = bp[2], y2b = bp[3];
        float sz = sqrtf(fmaxf((x2b - x1b) * (y2b - y1b), 0.0f));
        int lv = (int)floorf(4.0f + log2f(sz / 224.0f + 1e-8f));
        lv = lv < 2 ? 2 : (lv > 5 ? 5 : lv);
        const int l = lv - 2;
        const int   Ht[4] = {256, 128, 64, 32};
        const float St[4] = {0.25f, 0.125f, 0.0625f, 0.03125f};
        const int H = Ht[l], W = Ht[l];
        const float sc = St[l];
        const float x1f = x1b * sc - 0.5f, y1f = y1b * sc - 0.5f;
        const float x2f = x2b * sc - 0.5f, y2f = y2b * sc - 0.5f;
        const float bw = (x2f - x1f) * (1.0f / 7.0f);
        const float bh = (y2f - y1f) * (1.0f / 7.0f);
        int xmin = W - 1, xmax = 0;
        for (int k = 0; k < 14; ++k) {
            const float offk = (float)(k >> 1) + 0.25f + 0.5f * (float)(k & 1);
            // x samples -> Ax support
            float xs = x1f + offk * bw;
            float vx = (xs >= -1.0f && xs <= (float)W) ? 1.0f : 0.0f;
            float xc = fminf(fmaxf(xs, 0.0f), (float)(W - 1));
            float fx = floorf(xc);
            int   i0 = (int)fx;
            int   i1 = (i0 + 1 <= W - 1) ? i0 + 1 : W - 1;
            float lx = xc - fx;
            sXi[k][0] = i0;               sXi[k][1] = i1;
            sXw[k][0] = vx * 0.5f * (1.0f - lx);
            sXw[k][1] = vx * 0.5f * lx;
            if (i0 < xmin) xmin = i0;
            if (i1 > xmax) xmax = i1;
            // y samples -> per-by row weights (<=4 rows per output row)
            float ys = y1f + offk * bh;
            float vy = (ys >= -1.0f && ys <= (float)H) ? 1.0f : 0.0f;
            float yc = fminf(fmaxf(ys, 0.0f), (float)(H - 1));
            float fy = floorf(yc);
            int   j0 = (int)fy;
            int   j1 = (j0 + 1 <= H - 1) ? j0 + 1 : H - 1;
            float ly = yc - fy;
            const int byk = k >> 1, t = k & 1;
            sRy[byk][2 * t]     = j0;  sWy[byk][2 * t]     = vy * 0.5f * (1.0f - ly);
            sRy[byk][2 * t + 1] = j1;  sWy[byk][2 * t + 1] = vy * 0.5f * ly;
        }
        int Bb  = xmax - xmin + 1;
        int nch = (Bb + 31) >> 5;  if (nch > 4) nch = 4;
        sI[0] = l; sI[1] = H; sI[2] = W; sI[3] = H * W;
        sI[4] = xmin; sI[5] = (Bb < BMAX ? Bb : BMAX);
        sI[6] = nch;  sI[7] = (nch + 1) >> 1;
    }
    __syncthreads();

    const int l = sI[0], H = sI[1], W = sI[2], HW = sI[3];
    const int xmin = sI[4], Bc = sI[5], nchunk = sI[6], nslab = sI[7];
    (void)H;
    const float* fptr = (l == 0) ? f0 : (l == 1) ? f1 : (l == 2) ? f2 : f3;
    const float* fb   = fptr + (size_t)nb * 256 * (size_t)HW;

    // ---------------- build Ax (hi/lo f16) in LDS ----------------
    for (int i = tid; i < 16 * BMAX; i += 256) { sAxh[i] = (_Float16)0.0f; sAxl[i] = (_Float16)0.0f; }
    __syncthreads();
    if (tid < 7) {
        const int bin = tid;  // each bin owned by one thread -> race free
        for (int t = 0; t < 2; ++t) {
            const int k = 2 * bin + t;
            for (int d = 0; d < 2; ++d) {
                const int xr = sXi[k][d] - xmin;
                if (xr >= 0 && xr < BMAX) {
                    const int ii = bin * BMAX + xr;
                    float cur = (float)sAxh[ii] + (float)sAxl[ii] + sXw[k][d];
                    _Float16 h = (_Float16)cur;
                    sAxh[ii] = h;
                    sAxl[ii] = (_Float16)(cur - (float)h);
                }
            }
        }
    }
    __syncthreads();

    const int lane = tid & 31;
    const int wid  = tid >> 5;
    const int hsel = (lane >> 4) & 1;  // lane half (ISA 16-bit operand layouts)
    const int m    = lane & 15;

    // ---------------- main loop: 16 channel tiles of 16 ----------------
    for (int ct = 0; ct < 16; ++ct) {
        v8f acc;
        #pragma unroll
        for (int r = 0; r < 8; ++r) acc[r] = 0.0f;

        for (int s = 0; s < nslab; ++s) {
            // Stage 1: y-interp + y-pool into G slab (all 256 threads, coalesced x)
            for (int e = tid; e < 7 * 16 * SLAB; e += 256) {
                const int xl = e & (SLAB - 1);
                const int cc = (e >> 6) & 15;
                const int by = e >> 10;
                const int xg = s * SLAB + xl;
                float val = 0.0f;
                if (xg < Bc) {
                    const float* fc = fb + (size_t)(ct * 16 + cc) * HW + (xmin + xg);
                    #pragma unroll
                    for (int j = 0; j < 4; ++j)
                        val += sWy[by][j] * fc[sRy[by][j] * W];
                }
                const int gi = (by * 16 + cc) * SLAB + xl;
                _Float16 h = (_Float16)val;
                sGh[gi] = h;
                sGl[gi] = (_Float16)(val - (float)h);
            }
            __syncthreads();

            if (wid < 7) {
                // Stage 2: out[16c x 7bx] += G[16c x K] * Ax[K x 16] via WMMA
                const int by = wid;
                #pragma unroll
                for (int q = 0; q < 2; ++q) {
                    const int g = 2 * s + q;             // global 32-wide K chunk
                    if (g < nchunk) {
                        // A (16x32 f16): lane half hsel -> K bases 8*hsel and 8*hsel+16
                        const int arow = (by * 16 + m) * SLAB + 32 * q + 8 * hsel;
                        v8h ah0 = *(const v8h*)(sGh + arow);
                        v8h ah1 = *(const v8h*)(sGh + arow + 16);
                        v8h al0 = *(const v8h*)(sGl + arow);
                        v8h al1 = *(const v8h*)(sGl + arow + 16);
                        // B (32x16 f16): lane col m, K bases 8*hsel and 8*hsel+16
                        const int brow = m * BMAX + g * 32 + 8 * hsel;
                        v8h bh0 = *(const v8h*)(sAxh + brow);
                        v8h bh1 = *(const v8h*)(sAxh + brow + 16);
                        v8h bl0 = *(const v8h*)(sAxl + brow);
                        v8h bl1 = *(const v8h*)(sAxl + brow + 16);
                        v16h Ah, Al, Bh, Bl;
                        #pragma unroll
                        for (int i = 0; i < 8; ++i) {
                            Ah[i] = ah0[i]; Ah[i + 8] = ah1[i];
                            Al[i] = al0[i]; Al[i + 8] = al1[i];
                            Bh[i] = bh0[i]; Bh[i + 8] = bh1[i];
                            Bl[i] = bl0[i]; Bl[i + 8] = bl1[i];
                        }
                        // error-compensated f16 matmul: AhBh + AhBl + AlBh (~f32 acc.)
                        acc = __builtin_amdgcn_wmma_f32_16x16x32_f16(false, Ah, false, Bh, (short)0, acc, false, false);
                        acc = __builtin_amdgcn_wmma_f32_16x16x32_f16(false, Ah, false, Bl, (short)0, acc, false, false);
                        acc = __builtin_amdgcn_wmma_f32_16x16x32_f16(false, Al, false, Bh, (short)0, acc, false, false);
                    }
                }
            } else if (ct + 1 < 16) {
                // Wave 7: prefetch next channel tile's rows (global_prefetch_b8)
                for (int k2 = lane; k2 < 448; k2 += 32) {
                    const int by2 = k2 >> 6;
                    const int j2  = (k2 >> 4) & 3;
                    const int c2  = k2 & 15;
                    const float* p = fb + (size_t)((ct + 1) * 16 + c2) * HW
                                        + sRy[by2][j2] * W + xmin + s * SLAB;
                    __builtin_prefetch(p, 0, 1);
                }
            }
            __syncthreads();
        }

        // Store D tile: lane (hsel,m) vgpr r -> channel r+8*hsel, bin-x m
        if (wid < 7 && m < OUTS) {
            const int by = wid;
            #pragma unroll
            for (int r = 0; r < 8; ++r) {
                const int c = ct * 16 + r + 8 * hsel;
                out[(((size_t)roi * 256 + c) * 7 + by) * 7 + m] = acc[r];
            }
        }
    }
}

extern "C" void kernel_launch(void* const* d_in, const int* in_sizes, int n_in,
                              void* d_out, int out_size, void* d_ws, size_t ws_size,
                              hipStream_t stream) {
    (void)n_in; (void)out_size; (void)d_ws; (void)ws_size;
    const float* f0 = (const float*)d_in[0];
    const float* f1 = (const float*)d_in[1];
    const float* f2 = (const float*)d_in[2];
    const float* f3 = (const float*)d_in[3];
    const float* bx = (const float*)d_in[4];
    float* out = (float*)d_out;
    const int M = in_sizes[4] / 4;  // 1024 ROIs
    roi_align_fpn_kernel<<<M, 256, 0, stream>>>(f0, f1, f2, f3, bx, out);
}
